// YOLOv4_33938831573700
// MI455X (gfx1250) — matrix operand order repaired and checked by reading, not compiled, not run on hardware
//
#include <hip/hip_runtime.h>

// ---------------------------------------------------------------------------
// YOLOv4 inference for MI455X (gfx1250, wave32).
// Conv path: double-LDS-staged implicit-GEMM with V_WMMA_F32_16X16X32_F16.
//   - block (128 thr = 4 waves) owns a 64(Cout) x 128(pixel) supertile
//   - per K-step: weights (64x32, f32->f16) AND im2col activation tile
//     (32x128 f16) staged cooperatively into padded LDS
//   - each wave: 4M x 2N register blocking -> 8 WMMAs per K-step, A
//     fragments built with aligned ds_load_b128, B from 16-bit LDS reads
//   - fused bias + leaky-relu + residual-add + channel-offset (CSP concat)
// Tail: decode / top-k / per-class NMS.
// ---------------------------------------------------------------------------

typedef __attribute__((ext_vector_type(16))) _Float16 v16h;
typedef __attribute__((ext_vector_type(8)))  _Float16 v8h;
typedef __attribute__((ext_vector_type(8)))  float    v8f;

// ------------------------------ small kernels ------------------------------

__global__ __launch_bounds__(256) void cvt_f32_f16(const float* __restrict__ in,
                                                   _Float16* __restrict__ out, int n) {
  int i = blockIdx.x * blockDim.x + threadIdx.x;
  if (i < n) out[i] = (_Float16)in[i];
}

// dst[b,c,y,x] += src[b,c,y/2,x/2]   (FPN top-down up2 + add)
__global__ __launch_bounds__(256) void upadd_kernel(_Float16* __restrict__ dst,
                                                    const _Float16* __restrict__ src,
                                                    int C, int H, int W, int B) {
  int i = blockIdx.x * blockDim.x + threadIdx.x;
  int total = B * C * H * W;
  if (i >= total) return;
  int x = i % W;
  int y = (i / W) % H;
  int c = (i / (W * H)) % C;
  int b = i / (W * H * C);
  int hs = H >> 1, ws = W >> 1;
  float v = (float)dst[i] + (float)src[(((size_t)b * C + c) * hs + (y >> 1)) * ws + (x >> 1)];
  dst[i] = (_Float16)v;
}

// ------------------------- WMMA implicit-GEMM conv -------------------------
// GEMM view: M = Cout, N = pixels (per image), K = Cin*KS*KS.
// A (16x32 f16) VGPR layout: lane -> row M = lane%16;
//   element e<8  -> K = e + 8*(lane/16)
//   element e>=8 -> K = e + 8 + 8*(lane/16)
// B (32x16 f16): lane -> column N = lane%16 (output pixel); element e -> K row, same order as A.
// C/D (16x16 f32): acc[r] -> M = r + 8*(lane/16), N = lane%16.

#define MBLK 4        // 4 x 16 = 64 output channels per block
#define NBLK 2        // 2 x 16 = 32 pixels per wave (4 waves -> 128 px/block)
#define WSTR 40       // weight LDS row stride in halves (32 + 8 pad)
#define BSTR 136      // B-tile LDS row stride in halves (128 + 8 pad)

__global__ __launch_bounds__(128) void conv_wmma_kernel(
    const _Float16* __restrict__ in, int Cin, int Hin, int Win,
    const float* __restrict__ Wt, const float* __restrict__ Bs,
    int KS, int stride, int padb,
    _Float16* __restrict__ out, int Cout, int CT, int coff,
    const _Float16* __restrict__ resid, int act,
    int Hout, int Wout, int Mt, int Nt)
{
  __shared__ _Float16 wlds[64 * WSTR];   // weights: 64 M-rows x 32 K (padded)
  __shared__ _Float16 blds[32 * BSTR];   // im2col:  32 K-rows x 128 px (padded)

  int t    = threadIdx.x;
  int wid  = t >> 5;                     // wave 0..3
  int lane = t & 31;
  int half = lane >> 4;
  int lm   = lane & 15;

  // block -> (batch, m-supertile, n-supertile); uniform, no early exit
  int blk = blockIdx.x;
  int per = Mt * Nt;
  int b   = blk / per;
  int rm  = blk % per;
  int m0  = (rm / Nt) * 64;              // first output channel of block
  int pb0 = (rm % Nt) * 128;             // first pixel of block
  int nb0 = pb0 + wid * 32;              // first pixel of this wave

  int Np   = Hout * Wout;
  int KK   = KS * KS;
  int Ktot = Cin * KK;

  // ---- loader thread geometry (K-invariant, hoisted out of K loop) ----
  // weights: thread -> row r (16 contiguous K each)
  int wi   = t * 16;
  int wr   = wi >> 5;                    // 0..63
  int wk   = wi & 31;                    // 0 or 16
  int wcm  = m0 + wr;
  const float* wg = Wt + (size_t)(wcm < Cout ? wcm : 0) * Ktot;
  // im2col: thread -> K-row lrow, 32 contiguous pixels starting at px0
  int lrow = t >> 2;                     // 0..31
  int lch  = (t & 3) * 32;               // 0,32,64,96
  int px0  = pb0 + lch;
  int loy  = px0 / Wout;                 // single division, K-invariant
  int lox  = px0 - loy * Wout;

  v8f acc[MBLK][NBLK];
#pragma unroll
  for (int mi = 0; mi < MBLK; ++mi)
#pragma unroll
    for (int ni = 0; ni < NBLK; ++ni) acc[mi][ni] = (v8f){};

  for (int k0 = 0; k0 < Ktot; k0 += 32) {
    // ---- stage weights: 64 rows x 32 K, f32 -> f16 ----
    if (wcm < Cout && (k0 + 32) < Ktot) __builtin_prefetch(wg + k0 + 32 + wk, 0, 3);
#pragma unroll
    for (int j = 0; j < 16; ++j) {
      int kj = k0 + wk + j;
      float w = (wcm < Cout && kj < Ktot) ? wg[kj] : 0.f;
      wlds[wr * WSTR + wk + j] = (_Float16)w;
    }

    // ---- stage im2col tile: K-row lrow, 32 pixels, division-free walk ----
    {
      int kb = k0 + lrow;
      bool kok = (kb < Ktot);
      int cin = 0, kh = 0, kw = 0;
      if (kok) { cin = kb / KK; int r = kb % KK; kh = r / KS; kw = r % KS; }
      const _Float16* ip = in + ((size_t)b * Cin + cin) * Hin * Win;
      int oy = loy, ox = lox;
#pragma unroll
      for (int j = 0; j < 32; ++j) {
        float v = 0.f;
        if (kok && (px0 + j) < Np) {
          int iy = oy * stride - padb + kh;
          int ix = ox * stride - padb + kw;
          if ((unsigned)iy < (unsigned)Hin && (unsigned)ix < (unsigned)Win)
            v = (float)ip[(size_t)iy * Win + ix];
        }
        blds[lrow * BSTR + lch + j] = (_Float16)v;
        if (++ox == Wout) { ox = 0; ++oy; }
      }
    }
    __syncthreads();

    // ---- B fragments from LDS: lane = px column (lane%16), element e -> K row (table order) ----
    v16h bf[NBLK];
#pragma unroll
    for (int ni = 0; ni < NBLK; ++ni) {
      const _Float16* bp = &blds[(half * 8) * BSTR + wid * 32 + ni * 16 + lm];
#pragma unroll
      for (int e = 0; e < 8; ++e) { bf[ni][e] = bp[e * BSTR]; bf[ni][e + 8] = bp[(16 + e) * BSTR]; }
    }

    // ---- A fragments from LDS (layout matches ISA A-fragment exactly) ----
#pragma unroll
    for (int mi = 0; mi < MBLK; ++mi) {
      const v8h* lp = (const v8h*)&wlds[(mi * 16 + lm) * WSTR + half * 8];
      v8h lo = lp[0];                    // K = 8*half .. 8*half+7
      v8h hi = lp[2];                    // K = 16+8*half .. 23+8*half
      v16h a;
#pragma unroll
      for (int e = 0; e < 8; ++e) { a[e] = lo[e]; a[e + 8] = hi[e]; }
#pragma unroll
      for (int ni = 0; ni < NBLK; ++ni)
        acc[mi][ni] = __builtin_amdgcn_wmma_f32_16x16x32_f16(false, a, false, bf[ni],
                                                             (short)0, acc[mi][ni],
                                                             false, false);
    }
    __syncthreads();
  }

  // ---- epilogue: bias + leaky-relu + residual, write f16 with channel offset
#pragma unroll
  for (int ni = 0; ni < NBLK; ++ni) {
    int px = nb0 + ni * 16 + lm;         // this lane's output pixel (N = lane%16)
    if (px < Np) {
      int oy = px / Wout, ox = px % Wout;
#pragma unroll
      for (int mi = 0; mi < MBLK; ++mi) {
#pragma unroll
        for (int r = 0; r < 8; ++r) {
          int cl = m0 + mi * 16 + r + 8 * half;
          if (cl < Cout) {
            float v = acc[mi][ni][r] + Bs[cl];
            if (act) v = (v > 0.f) ? v : 0.1f * v;
            if (resid) v += (float)resid[(((size_t)b * Cout + cl) * Hout + oy) * Wout + ox];
            out[(((size_t)b * CT + coff + cl) * Hout + oy) * Wout + ox] = (_Float16)v;
          }
        }
      }
    }
  }
}

// ------------------------------- decode ------------------------------------

__global__ __launch_bounds__(256) void decode_kernel(
    const _Float16* __restrict__ y3, const _Float16* __restrict__ y4,
    const _Float16* __restrict__ y5, const int* __restrict__ oshape,
    float* __restrict__ boxes, float* __restrict__ score, float* __restrict__ clsf)
{
  const int Ntot = 3 * (52 * 52 + 26 * 26 + 13 * 13);   // 10647
  int gid = blockIdx.x * blockDim.x + threadIdx.x;
  if (gid >= 8 * Ntot) return;
  int b = gid / Ntot, n = gid % Ntot;
  const int n0 = 52 * 52 * 3, n1 = n0 + 26 * 26 * 3;
  const _Float16* y; int Wd, strd, si, nn;
  if (n < n0)      { si = 0; y = y3; Wd = 52; strd = 8;  nn = n; }
  else if (n < n1) { si = 1; y = y4; Wd = 26; strd = 16; nn = n - n0; }
  else             { si = 2; y = y5; Wd = 13; strd = 32; nn = n - n1; }
  const float anch[3][3][2] = {{{12,16},{19,36},{40,28}},
                               {{36,75},{76,55},{72,146}},
                               {{142,110},{192,243},{459,401}}};
  int a = nn % 3, pos = nn / 3;
  int yy = pos / Wd, xx = pos % Wd;
  size_t plane = (size_t)Wd * Wd;
  const _Float16* p = y + ((size_t)b * 255 + a * 85) * plane + (size_t)yy * Wd + xx;

  float tx = (float)p[0],          ty = (float)p[plane];
  float tw = (float)p[2 * plane],  th = (float)p[3 * plane];
  float to = (float)p[4 * plane];
  float cx = (1.f / (1.f + __expf(-tx)) + (float)xx) * (float)strd;
  float cy = (1.f / (1.f + __expf(-ty)) + (float)yy) * (float)strd;
  tw = fminf(fmaxf(tw, -9.f), 9.f);
  th = fminf(fmaxf(th, -9.f), 9.f);
  float bw = __expf(tw) * anch[si][a][0];
  float bh = __expf(th) * anch[si][a][1];
  float obj = 1.f / (1.f + __expf(-to));

  float best = -1e30f; int bc = 0;
  for (int c = 0; c < 80; ++c) {
    float v = (float)p[(size_t)(5 + c) * plane];
    if (v > best) { best = v; bc = c; }
  }
  float conf = obj * (1.f / (1.f + __expf(-best)));   // sigmoid monotone -> argmax safe

  float hr = (float)oshape[0] / 416.f, wr = (float)oshape[1] / 416.f;
  float x1 = (cx - bw * 0.5f) * wr, y1c = (cy - bh * 0.5f) * hr;
  float x2 = (cx + bw * 0.5f) * wr, y2c = (cy + bh * 0.5f) * hr;
  float sc = (conf >= 0.25f) ? conf : 0.f;

  size_t o = (size_t)b * Ntot + n;
  boxes[o * 4 + 0] = x1; boxes[o * 4 + 1] = y1c;
  boxes[o * 4 + 2] = x2; boxes[o * 4 + 3] = y2c;
  score[o] = sc; clsf[o] = (float)bc;
}

// --------------------------- top-k (300) per image -------------------------

__global__ __launch_bounds__(256) void topk_kernel(
    float* __restrict__ score, const float* __restrict__ boxes,
    const float* __restrict__ clsf,
    float* __restrict__ tb, float* __restrict__ ts, float* __restrict__ tc)
{
  const int N = 3 * (52 * 52 + 26 * 26 + 13 * 13);
  int b = blockIdx.x, t = threadIdx.x;
  score += (size_t)b * N; boxes += (size_t)b * N * 4; clsf += (size_t)b * N;
  tb += b * 300 * 4; ts += b * 300; tc += b * 300;
  __shared__ float sv[256];
  __shared__ int   si[256];
  for (int it = 0; it < 300; ++it) {
    float best = -1e30f; int bi = 0;
    for (int i = t; i < N; i += 256) {
      float v = score[i];
      if (v > best) { best = v; bi = i; }
    }
    sv[t] = best; si[t] = bi;
    __syncthreads();
    for (int s = 128; s > 0; s >>= 1) {
      if (t < s && sv[t + s] > sv[t]) { sv[t] = sv[t + s]; si[t] = si[t + s]; }
      __syncthreads();
    }
    if (t == 0) {
      int id = si[0]; float v = sv[0];
      ts[it] = (v > 0.f) ? v : 0.f;
      tb[it * 4 + 0] = boxes[id * 4 + 0]; tb[it * 4 + 1] = boxes[id * 4 + 1];
      tb[it * 4 + 2] = boxes[id * 4 + 2]; tb[it * 4 + 3] = boxes[id * 4 + 3];
      tc[it] = clsf[id];
      score[id] = -1e30f;                 // consume (decode rewrites next call)
    }
    __syncthreads();
  }
}

// ------------------------ per-class NMS + write out ------------------------

__global__ __launch_bounds__(320) void nms_kernel(
    const float* __restrict__ tb, const float* __restrict__ ts,
    const float* __restrict__ tc, float* __restrict__ out)
{
  int b = blockIdx.x, t = threadIdx.x;
  tb += b * 1200; ts += b * 300; tc += b * 300; out += b * 1800;
  __shared__ float bxs[1200], sc[300], cl[300];
  __shared__ int keep[300];
  __shared__ int sup;
  for (int i = t; i < 300; i += 320) {
    bxs[i * 4 + 0] = tb[i * 4 + 0]; bxs[i * 4 + 1] = tb[i * 4 + 1];
    bxs[i * 4 + 2] = tb[i * 4 + 2]; bxs[i * 4 + 3] = tb[i * 4 + 3];
    sc[i] = ts[i]; cl[i] = tc[i]; keep[i] = 0;
  }
  __syncthreads();
  for (int i = 0; i < 300; ++i) {
    if (t == 0) sup = 0;
    __syncthreads();
    if (t < i && keep[t] && cl[t] == cl[i]) {   // class-offset trick == same-class IoU
      float ax1 = bxs[i * 4], ay1 = bxs[i * 4 + 1], ax2 = bxs[i * 4 + 2], ay2 = bxs[i * 4 + 3];
      float bx1 = bxs[t * 4], by1 = bxs[t * 4 + 1], bx2 = bxs[t * 4 + 2], by2 = bxs[t * 4 + 3];
      float aA = fmaxf(ax2 - ax1, 0.f) * fmaxf(ay2 - ay1, 0.f);
      float aB = fmaxf(bx2 - bx1, 0.f) * fmaxf(by2 - by1, 0.f);
      float ix1 = fmaxf(ax1, bx1), iy1 = fmaxf(ay1, by1);
      float ix2 = fminf(ax2, bx2), iy2 = fminf(ay2, by2);
      float inter = fmaxf(ix2 - ix1, 0.f) * fmaxf(iy2 - iy1, 0.f);
      float iou = inter / (aA + aB - inter + 1e-7f);
      if (iou > 0.4f) atomicOr(&sup, 1);
    }
    __syncthreads();
    if (t == 0) keep[i] = (sup == 0) && (sc[i] > 0.f);
    __syncthreads();
  }
  for (int i = t; i < 300; i += 320) {
    float k = keep[i] ? 1.f : 0.f;
    out[i * 6 + 0] = k * bxs[i * 4 + 0]; out[i * 6 + 1] = k * bxs[i * 4 + 1];
    out[i * 6 + 2] = k * bxs[i * 4 + 2]; out[i * 6 + 3] = k * bxs[i * 4 + 3];
    out[i * 6 + 4] = k * sc[i];          out[i * 6 + 5] = k * cl[i];
  }
}

// ------------------------------ host launcher ------------------------------

struct CP { const float* b; const float* w; };
struct CSPP { CP down, out, p1, p2; CP res[8][2]; int n; };

extern "C" void kernel_launch(void* const* d_in, const int* in_sizes, int n_in,
                              void* d_out, int out_size, void* d_ws, size_t ws_size,
                              hipStream_t stream) {
  (void)in_sizes; (void)n_in; (void)out_size; (void)ws_size;
  const int B = 8;
  const float* x_f32 = (const float*)d_in[0];
  const int* oshape  = (const int*)d_in[1];

  // ---- params: jax pytree order (dict keys sorted; conv dict = b then w) ----
  int ii = 2;
  auto getCP = [&]() { CP c; c.b = (const float*)d_in[ii++]; c.w = (const float*)d_in[ii++]; return c; };
  auto getCSP = [&](int n) {
    CSPP s; s.n = n;
    s.down = getCP(); s.out = getCP(); s.p1 = getCP(); s.p2 = getCP();
    for (int i = 0; i < n; ++i) { s.res[i][0] = getCP(); s.res[i][1] = getCP(); }
    return s;
  };
  CSPP cs1 = getCSP(1), cs2 = getCSP(2), cs3 = getCSP(8), cs4 = getCSP(8), cs5 = getCSP(4);
  CP lat3 = getCP(), lat4 = getCP(), lat5 = getCP();
  CP sm3 = getCP(), sm4 = getCP(), sm5 = getCP();
  CP down34 = getCP(), down45 = getCP();
  CP yl0 = getCP(), yl1 = getCP(), yl2 = getCP();

  // ---- workspace bump allocator ----
  char* base = (char*)d_ws;
  size_t off = 0;
  auto ah = [&](size_t n) { _Float16* p = (_Float16*)(base + off);
                            off = (off + n * sizeof(_Float16) + 255) & ~(size_t)255; return p; };
  auto af = [&](size_t n) { float* p = (float*)(base + off);
                            off = (off + n * sizeof(float) + 255) & ~(size_t)255; return p; };

  // persistent (live across backbone -> FPN -> tail)
  _Float16* O3 = ah((size_t)B * 256 * 52 * 52);
  _Float16* O4 = ah((size_t)B * 512 * 26 * 26);
  _Float16* O5 = ah((size_t)B * 1024 * 13 * 13);
  float* BOX = af((size_t)B * 10647 * 4);
  float* SCR = af((size_t)B * 10647);
  float* CLS = af((size_t)B * 10647);
  float* TBB = af((size_t)B * 300 * 4);
  float* TSS = af((size_t)B * 300);
  float* TCC = af((size_t)B * 300);

  size_t poolStart = off;
  // transient backbone pool
  _Float16* XH  = ah((size_t)B * 3 * 416 * 416);
  _Float16* D   = ah((size_t)B * 64 * 208 * 208);
  _Float16* CAT = ah((size_t)B * 64 * 208 * 208);
  _Float16* R0  = ah((size_t)B * 32 * 208 * 208);
  _Float16* R1  = ah((size_t)B * 32 * 208 * 208);
  _Float16* RT  = ah((size_t)B * 32 * 208 * 208);
  _Float16* O1  = ah((size_t)B * 64 * 208 * 208);
  _Float16* O2  = ah((size_t)B * 128 * 104 * 104);

  // FPN/head pool aliases the (then-dead) backbone pool
  off = poolStart;
  _Float16* P3  = ah((size_t)B * 256 * 52 * 52);
  _Float16* P3S = ah((size_t)B * 256 * 52 * 52);
  _Float16* P4  = ah((size_t)B * 256 * 26 * 26);
  _Float16* P4S = ah((size_t)B * 256 * 26 * 26);
  _Float16* P5  = ah((size_t)B * 256 * 13 * 13);
  _Float16* P5S = ah((size_t)B * 256 * 13 * 13);
  _Float16* N4  = ah((size_t)B * 256 * 26 * 26);
  _Float16* N5  = ah((size_t)B * 256 * 13 * 13);
  _Float16* Y3  = ah((size_t)B * 255 * 52 * 52);
  _Float16* Y4  = ah((size_t)B * 255 * 26 * 26);
  _Float16* Y5  = ah((size_t)B * 255 * 13 * 13);

  // ---- conv launcher: block = 64 Cout x 128 pixels ----
  auto conv = [&](const _Float16* in, int Cin, int Hin, int Win, CP p, int KS, int stride,
                  _Float16* outp, int Cout, int CT, int coff, const _Float16* resid, int act) {
    int Hout = Hin / stride, Wout = Win / stride;
    int padb = (KS == 3 && stride == 1) ? 1 : 0;   // jax SAME: s2/k3 pads (0,1), s1/k3 pads (1,1)
    int Mt = (Cout + 63) / 64, Nt = (Hout * Wout + 127) / 128;
    int blocks = B * Mt * Nt;
    conv_wmma_kernel<<<blocks, 128, 0, stream>>>(in, Cin, Hin, Win, p.w, p.b, KS, stride, padb,
                                                 outp, Cout, CT, coff, resid, act,
                                                 Hout, Wout, Mt, Nt);
  };

  // ---- CSP stage ----
  auto csp = [&](const _Float16* x, int Cin, int Hin, CSPP& s, int Cout, _Float16* outbuf) {
    int h = Cout / 2, H = Hin / 2;
    conv(x, Cin, Hin, Hin, s.down, 3, 2, D, Cout, Cout, 0, nullptr, 1);
    conv(D, Cout, H, H, s.p1, 1, 1, CAT, h, Cout, 0, nullptr, 1);      // a -> concat[0,h)
    conv(D, Cout, H, H, s.p2, 1, 1, R0, h, h, 0, nullptr, 1);          // b
    _Float16* bcur = R0; _Float16* bnext = R1;
    for (int i = 0; i < s.n; ++i) {
      conv(bcur, h, H, H, s.res[i][0], 1, 1, RT, h, h, 0, nullptr, 1); // t = act(1x1(b))
      if (i == s.n - 1) {
        conv(RT, h, H, H, s.res[i][1], 3, 1, CAT, h, Cout, h, bcur, 1); // b + act(3x3(t)) -> concat[h,2h)
      } else {
        conv(RT, h, H, H, s.res[i][1], 3, 1, bnext, h, h, 0, bcur, 1);
        _Float16* tmp = bcur; bcur = bnext; bnext = tmp;
      }
    }
    conv(CAT, Cout, H, H, s.out, 1, 1, outbuf, Cout, Cout, 0, nullptr, 1);
  };

  // ---- pipeline ----
  int nx = B * 3 * 416 * 416;
  cvt_f32_f16<<<(nx + 255) / 256, 256, 0, stream>>>(x_f32, XH, nx);

  csp(XH, 3, 416, cs1, 64, O1);
  csp(O1, 64, 208, cs2, 128, O2);
  csp(O2, 128, 104, cs3, 256, O3);
  csp(O3, 256, 52, cs4, 512, O4);
  csp(O4, 512, 26, cs5, 1024, O5);

  // FPN
  conv(O5, 1024, 13, 13, lat5, 1, 1, P5, 256, 256, 0, nullptr, 1);
  conv(O4, 512, 26, 26, lat4, 1, 1, P4, 256, 256, 0, nullptr, 1);
  { int tot = B * 256 * 26 * 26;
    upadd_kernel<<<(tot + 255) / 256, 256, 0, stream>>>(P4, P5, 256, 26, 26, B); }
  conv(O3, 256, 52, 52, lat3, 1, 1, P3, 256, 256, 0, nullptr, 1);
  { int tot = B * 256 * 52 * 52;
    upadd_kernel<<<(tot + 255) / 256, 256, 0, stream>>>(P3, P4, 256, 52, 52, B); }
  conv(P3, 256, 52, 52, sm3, 3, 1, P3S, 256, 256, 0, nullptr, 1);
  conv(P4, 256, 26, 26, sm4, 3, 1, P4S, 256, 256, 0, nullptr, 1);
  conv(P5, 256, 13, 13, sm5, 3, 1, P5S, 256, 256, 0, nullptr, 1);

  // PAN (n3 = P3S); resid pointer implements "+ p4 / + p5" after activation
  conv(P3S, 256, 52, 52, down34, 3, 2, N4, 256, 256, 0, P4S, 1);
  conv(N4,  256, 26, 26, down45, 3, 2, N5, 256, 256, 0, P5S, 1);

  // heads (linear)
  conv(P3S, 256, 52, 52, yl0, 1, 1, Y3, 255, 255, 0, nullptr, 0);
  conv(N4,  256, 26, 26, yl1, 1, 1, Y4, 255, 255, 0, nullptr, 0);
  conv(N5,  256, 13, 13, yl2, 1, 1, Y5, 255, 255, 0, nullptr, 0);

  // decode -> top-k -> NMS -> d_out [8,300,6]
  int np = B * 10647;
  decode_kernel<<<(np + 255) / 256, 256, 0, stream>>>(Y3, Y4, Y5, oshape, BOX, SCR, CLS);
  topk_kernel<<<B, 256, 0, stream>>>(SCR, BOX, CLS, TBB, TSS, TCC);
  nms_kernel<<<B, 320, 0, stream>>>(TBB, TSS, TCC, (float*)d_out);
}
